// EfficientAttention_86363202388614
// MI455X (gfx1250) — compile-verified, matrix operand order checked by reading, not run on hardware
//
#include <hip/hip_runtime.h>
#include <hip/hip_bf16.h>

typedef __attribute__((ext_vector_type(16))) __bf16 v16bf;
typedef __attribute__((ext_vector_type(8)))  float  v8f;

constexpr int B_  = 4;
constexpr int C_  = 256;
constexpr int CK_ = 32;
constexpr int N_  = 16384;   // T*H*W = 16*32*32

// 16-bit A/B fragment element -> K index mapping (CDNA5 ISA 7.12.2, 16x32 A):
// lane L: row = L&15, kbase = (L>>4)*8 ; elem i: K = (i>>3)*16 + kbase + (i&7)
__device__ __forceinline__ int fragK(int lane, int i) {
    return ((i >> 3) << 4) + ((lane >> 4) << 3) + (i & 7);
}

// ---------------------------------------------------------------------------
// K1: logits L[b,k,n] = Wk[k,:] . ref[b,:,n] + bk[k]
// grid (N/64, 2 mtiles, B), block 128 (4 waves, each one 16x16 n-tile)
// ---------------------------------------------------------------------------
__global__ void ea_k1_logits(const float* __restrict__ ref,
                             const float* __restrict__ Wk,
                             const float* __restrict__ bk,
                             float* __restrict__ L) {
    __shared__ float lds_w[16 * 256];
    const int b = blockIdx.z, mtile = blockIdx.y;
    const int w = threadIdx.x >> 5, lane = threadIdx.x & 31;
    const int mA = lane & 15;

    for (int idx = threadIdx.x; idx < 16 * 256; idx += 128) {
        int r = idx >> 8, c = idx & 255;
        lds_w[idx] = Wk[(mtile * 16 + r) * C_ + c];
    }
    __syncthreads();

    const int n0 = blockIdx.x * 64 + w * 16;
    v8f acc = {};
    for (int c0 = 0; c0 < C_; c0 += 32) {
        v16bf a, bb;
        #pragma unroll
        for (int i = 0; i < 16; ++i) {
            int K = fragK(lane, i);
            a[i]  = (__bf16)lds_w[mA * 256 + c0 + K];
            bb[i] = (__bf16)ref[((size_t)(b * C_ + c0 + K)) * N_ + n0 + mA];
        }
        acc = __builtin_amdgcn_wmma_f32_16x16x32_bf16(false, a, false, bb,
                                                      (short)0, acc, false, false);
    }
    #pragma unroll
    for (int r = 0; r < 8; ++r) {
        int kRow = mtile * 16 + r + ((lane >> 4) << 3);
        L[(size_t)(b * CK_ + kRow) * N_ + n0 + mA] = acc[r] + bk[kRow];
    }
}

// ---------------------------------------------------------------------------
// K2: per-row (b*CK rows) max + sum(exp); also zero S accumulator
// grid 128, block 256
// ---------------------------------------------------------------------------
__global__ void ea_k2_rowstats(const float* __restrict__ L,
                               float* __restrict__ mbuf,
                               float* __restrict__ zbuf,
                               float* __restrict__ Sbuf) {
    __shared__ float red[256];
    const int row = blockIdx.x;
    const size_t base = (size_t)row * N_;
    const int tid = threadIdx.x;

    float mx = -3.4e38f;
    for (int i = tid; i < N_; i += 256) mx = fmaxf(mx, L[base + i]);
    red[tid] = mx; __syncthreads();
    for (int s = 128; s > 0; s >>= 1) {
        if (tid < s) red[tid] = fmaxf(red[tid], red[tid + s]);
        __syncthreads();
    }
    const float m = red[0]; __syncthreads();

    float sum = 0.f;
    for (int i = tid; i < N_; i += 256) sum += __expf(L[base + i] - m);
    red[tid] = sum; __syncthreads();
    for (int s = 128; s > 0; s >>= 1) {
        if (tid < s) red[tid] += red[tid + s];
        __syncthreads();
    }
    if (tid == 0) { mbuf[row] = m; zbuf[row] = red[0]; }
    Sbuf[(size_t)row * C_ + tid] = 0.f;   // zero accumulator for K3
}

// ---------------------------------------------------------------------------
// K3: S[b,k,c] += sum_n exp(L[b,k,n]-m[b,k]) * ref[b,c,n]   (unnormalized KR)
// grid (8 k-parts, C/16 ctiles, B*2 mtiles), block 128 (4 waves x 512 n each)
// ---------------------------------------------------------------------------
__global__ void ea_k3_kr(const float* __restrict__ L,
                         const float* __restrict__ mbuf,
                         const float* __restrict__ ref,
                         float* __restrict__ Sbuf) {
    __shared__ float lds_a[4][16][33];
    __shared__ float lds_b[4][16][33];
    const int w = threadIdx.x >> 5, lane = threadIdx.x & 31;
    const int kpart = blockIdx.x, ctile = blockIdx.y;
    const int b = blockIdx.z >> 1, mtile = blockIdx.z & 1;
    const int mA = lane & 15;
    const int rowk0 = b * CK_ + mtile * 16;

    v8f acc = {};
    const int nw0 = kpart * 2048 + w * 512;
    for (int it = 0; it < 16; ++it) {
        const int n0 = nw0 + it * 32;
        #pragma unroll
        for (int r = 0; r < 16; ++r) {
            lds_a[w][r][lane] = __expf(L[(size_t)(rowk0 + r) * N_ + n0 + lane]
                                       - mbuf[rowk0 + r]);
            lds_b[w][r][lane] = ref[((size_t)(b * C_ + ctile * 16 + r)) * N_ + n0 + lane];
        }
        // wave-private LDS region; intra-wave DS ordering handled by s_wait_dscnt
        v16bf a, bb;
        #pragma unroll
        for (int i = 0; i < 16; ++i) {
            int K = fragK(lane, i);
            a[i]  = (__bf16)lds_a[w][mA][K];
            bb[i] = (__bf16)lds_b[w][mA][K];
        }
        acc = __builtin_amdgcn_wmma_f32_16x16x32_bf16(false, a, false, bb,
                                                      (short)0, acc, false, false);
    }
    #pragma unroll
    for (int r = 0; r < 8; ++r) {
        int kRow = rowk0 + r + ((lane >> 4) << 3);
        atomicAdd(&Sbuf[(size_t)kRow * C_ + ctile * 16 + mA], acc[r]);
    }
}

// ---------------------------------------------------------------------------
// K4: ctx[b,k,v] = Wv[v,:] . (S[b,k,:]/Z[b,k]) + bv[v]   (tiny GEMM)
// grid 128 (= B*CK), block 256
// ---------------------------------------------------------------------------
__global__ void ea_k4_ctx(const float* __restrict__ Sbuf,
                          const float* __restrict__ zbuf,
                          const float* __restrict__ Wv,
                          const float* __restrict__ bv,
                          float* __restrict__ ctx) {
    __shared__ float srow[256];
    const int row = blockIdx.x;
    const float invZ = 1.0f / zbuf[row];
    srow[threadIdx.x] = Sbuf[(size_t)row * C_ + threadIdx.x] * invZ;
    __syncthreads();
    const int v = threadIdx.x;
    float acc = bv[v];
    for (int c = 0; c < C_; ++c) acc += Wv[v * C_ + c] * srow[c];
    ctx[(size_t)row * C_ + v] = acc;
}

// ---------------------------------------------------------------------------
// K5: fused  q = softmax_k(Wq.src+bq);  out = gamma * ctx^T q + src
// grid (N/64, B), block 256 (8 waves). Static LDS ~62.9KB.
// ---------------------------------------------------------------------------
__global__ void ea_k5_out(const float* __restrict__ src,
                          const float* __restrict__ Wq,
                          const float* __restrict__ bq,
                          const float* __restrict__ ctx,
                          const float* __restrict__ gamma,
                          float* __restrict__ out) {
    __shared__ __bf16 lds_src[256][66];
    __shared__ __bf16 lds_ctx[32][258];
    __shared__ float  lds_q[32][66];
    __shared__ __bf16 lds_qb[32][66];
    const int b = blockIdx.y;
    const int n0 = blockIdx.x * 64;
    const int tid = threadIdx.x;
    const int w = tid >> 5, lane = tid & 31, mA = lane & 15;

    for (int idx = tid; idx < CK_ * C_; idx += 256) {
        int k = idx >> 8, v = idx & 255;
        lds_ctx[k][v] = (__bf16)ctx[(size_t)(b * CK_ + k) * C_ + v];
    }
    for (int idx = tid; idx < C_ * 64; idx += 256) {
        int c = idx >> 6, j = idx & 63;
        lds_src[c][j] = (__bf16)src[((size_t)(b * C_ + c)) * N_ + n0 + j];
    }
    __syncthreads();

    // --- query GEMM: wave w -> (mtile = w>>2, ntile = w&3) ---
    {
        const int mtile = w >> 2, ntile = w & 3;
        v8f acc = {};
        for (int c0 = 0; c0 < C_; c0 += 32) {
            v16bf a, bb;
            #pragma unroll
            for (int i = 0; i < 16; ++i) {
                int K = fragK(lane, i);
                a[i]  = (__bf16)Wq[(mtile * 16 + mA) * C_ + c0 + K];
                bb[i] = lds_src[c0 + K][ntile * 16 + mA];
            }
            acc = __builtin_amdgcn_wmma_f32_16x16x32_bf16(false, a, false, bb,
                                                          (short)0, acc, false, false);
        }
        #pragma unroll
        for (int r = 0; r < 8; ++r) {
            int k = mtile * 16 + r + ((lane >> 4) << 3);
            lds_q[k][ntile * 16 + mA] = acc[r] + bq[k];
        }
    }
    __syncthreads();

    // --- softmax over k (32) per column ---
    if (tid < 64) {
        float mx = -3.4e38f;
        #pragma unroll
        for (int k = 0; k < 32; ++k) mx = fmaxf(mx, lds_q[k][tid]);
        float e[32], s = 0.f;
        #pragma unroll
        for (int k = 0; k < 32; ++k) { e[k] = __expf(lds_q[k][tid] - mx); s += e[k]; }
        float inv = 1.0f / s;
        #pragma unroll
        for (int k = 0; k < 32; ++k) lds_qb[k][tid] = (__bf16)(e[k] * inv);
    }
    __syncthreads();

    // --- attention: out[v,n] = gamma * sum_k ctx[k,v] * q[k,n] + src[v,n] ---
    const float g = gamma[0];
    for (int t = 0; t < 8; ++t) {
        const int tile = w + t * 8;
        const int vtile = tile & 15, ntile = tile >> 4;
        v16bf a, bb;
        #pragma unroll
        for (int i = 0; i < 16; ++i) {
            int K = fragK(lane, i);      // K runs over k (=32 exactly one WMMA)
            a[i]  = lds_ctx[K][vtile * 16 + mA];
            bb[i] = lds_qb[K][ntile * 16 + mA];
        }
        v8f acc = {};
        acc = __builtin_amdgcn_wmma_f32_16x16x32_bf16(false, a, false, bb,
                                                      (short)0, acc, false, false);
        #pragma unroll
        for (int r = 0; r < 8; ++r) {
            int v = vtile * 16 + r + ((lane >> 4) << 3);
            size_t off = ((size_t)(b * C_ + v)) * N_ + n0 + ntile * 16 + mA;
            out[off] = g * acc[r] + src[off];
        }
    }
}

// ---------------------------------------------------------------------------
extern "C" void kernel_launch(void* const* d_in, const int* in_sizes, int n_in,
                              void* d_out, int out_size, void* d_ws, size_t ws_size,
                              hipStream_t stream) {
    const float* src   = (const float*)d_in[0];
    const float* ref   = (const float*)d_in[1];
    const float* Wq    = (const float*)d_in[2];
    const float* bq    = (const float*)d_in[3];
    const float* Wk    = (const float*)d_in[4];
    const float* bk    = (const float*)d_in[5];
    const float* Wv    = (const float*)d_in[6];
    const float* bv    = (const float*)d_in[7];
    const float* gamma = (const float*)d_in[8];
    float* out = (float*)d_out;

    float* Lbuf   = (float*)d_ws;                       // B*CK*N  = 8 MiB
    float* mbuf   = Lbuf + (size_t)B_ * CK_ * N_;       // 128
    float* zbuf   = mbuf + B_ * CK_;                    // 128
    float* Sbuf   = zbuf + B_ * CK_;                    // B*CK*C = 32768
    float* ctxbuf = Sbuf + (size_t)B_ * CK_ * C_;       // B*CK*C = 32768

    ea_k1_logits<<<dim3(N_ / 64, 2, B_), 128, 0, stream>>>(ref, Wk, bk, Lbuf);
    ea_k2_rowstats<<<B_ * CK_, 256, 0, stream>>>(Lbuf, mbuf, zbuf, Sbuf);
    ea_k3_kr<<<dim3(8, C_ / 16, B_ * 2), 128, 0, stream>>>(Lbuf, mbuf, ref, Sbuf);
    ea_k4_ctx<<<B_ * CK_, 256, 0, stream>>>(Sbuf, zbuf, Wv, bv, ctxbuf);
    ea_k5_out<<<dim3(N_ / 64, B_), 256, 0, stream>>>(src, Wq, bq, ctxbuf, gamma, out);
}
